// TransformerBlock_37641093382388
// MI455X (gfx1250) — matrix operand order repaired and checked by reading, not compiled, not run on hardware
//
#include <hip/hip_runtime.h>
#include <hip/hip_bf16.h>

typedef __bf16 v16bf  __attribute__((ext_vector_type(16)));
typedef __bf16 bf16x8 __attribute__((ext_vector_type(8)));
typedef float  v8f    __attribute__((ext_vector_type(8)));
typedef unsigned int u32x4 __attribute__((ext_vector_type(4)));
typedef int          i32x8 __attribute__((ext_vector_type(8)));
typedef int          i32x4 __attribute__((ext_vector_type(4)));

#define DM   1024
#define NH   16
#define HD   64
#define NE   8
#define TOPK 2
#define DFF  4096
#define BB   4
#define TT   1024
#define NTOK    (BB*TT)        // 4096 tokens
#define NASSIGN (NTOK*TOPK)    // 8192 expert-row assignments

#if defined(__gfx1250__) && __has_builtin(__builtin_amdgcn_tensor_load_to_lds) && __has_builtin(__builtin_amdgcn_s_wait_tensorcnt)
#define USE_TDM 1
#else
#define USE_TDM 0
#endif

// ---------- bf16 helpers (RNE) ----------
__device__ __forceinline__ __bf16 f2bf(float f) {
  unsigned int u = __float_as_uint(f);
  unsigned int r = u + 0x7FFFu + ((u >> 16) & 1u);
  unsigned short h = (unsigned short)(r >> 16);
  return __builtin_bit_cast(__bf16, h);
}
__device__ __forceinline__ float bf2f(__bf16 b) {
  unsigned short h = __builtin_bit_cast(unsigned short, b);
  return __uint_as_float(((unsigned int)h) << 16);
}
// combine two contiguous 16B halves into a 32B WMMA fragment
__device__ __forceinline__ v16bf frag2(const __bf16* p0, const __bf16* p1) {
  bf16x8 lo = *(const bf16x8*)p0;
  bf16x8 hi = *(const bf16x8*)p1;
  return __builtin_shufflevector(lo, hi, 0,1,2,3,4,5,6,7,8,9,10,11,12,13,14,15);
}

#if USE_TDM
// Issue a TDM 2D tile load: 8 rows x 1KB, row stride = rowStrideBytes,
// landing contiguously at LDS offset lds_off. D# packed per CDNA5 ISA 8.3/8.4.
__device__ __forceinline__ void tdm_load_b(unsigned lds_off, const __bf16* gsrc,
                                           unsigned rowStrideBytes) {
  unsigned long long ga = (unsigned long long)(size_t)gsrc;
  unsigned strideUnits = rowStrideBytes >> 3;        // 8-byte units
  u32x4 g0;
  g0[0] = 1u;                                        // count=1, user descriptor
  g0[1] = lds_off;                                   // lds_addr
  g0[2] = (unsigned)(ga & 0xFFFFFFFFu);              // global_addr[31:0]
  g0[3] = (unsigned)((ga >> 32) & 0x1FFFFFFu) | (2u << 30);  // addr[56:32] | type=2
  i32x8 g1;
  g1[0] = (3 << 16);                                 // data_size=3 (8B units)
  g1[1] = (int)((128u & 0xFFFFu) << 16);             // tensor_dim0[15:0]=128 @ bits 63:48
  g1[2] = (int)((8u) << 16);                         // tensor_dim0[31:16]=0 | tensor_dim1[15:0]=8
  g1[3] = (int)((128u) << 16);                       // tensor_dim1[31:16]=0 | tile_dim0=128
  g1[4] = 8;                                         // tile_dim1=8 | tile_dim2=0
  g1[5] = (int)strideUnits;                          // tensor_dim0_stride[31:0]
  g1[6] = 0;                                         // stride[47:32]=0 | dim1_stride[15:0]=0
  g1[7] = 0;
  i32x4 gz = {0, 0, 0, 0};
#if __clang_major__ >= 23
  i32x8 gz8 = {0, 0, 0, 0, 0, 0, 0, 0};
  __builtin_amdgcn_tensor_load_to_lds(g0, g1, gz, gz, gz8, 0);
#else
  __builtin_amdgcn_tensor_load_to_lds(g0, g1, gz, gz, 0);
#endif
}
#endif

// ---------- weight pack: f32 [K x N] -> bf16 WMMA-B fragment chunks ----------
// chunk(nt16, k32) of 512 bf16 at ((nt16*(K/32))+k32)*512; entry [lane][i]:
//   k = k32*32 + (lane>>4)*16 + i ; n = nt16*16 + (lane&15)
__global__ __launch_bounds__(256) void k_pack_b(const float* __restrict__ src,
                                                __bf16* __restrict__ dst, int K, int N) {
  size_t ebase = (size_t)blockIdx.y * K * N;
  int k32max = K >> 5;
  long long total = (long long)K * N;
  long long i = (long long)blockIdx.x * 256 + threadIdx.x;
  long long stride = (long long)gridDim.x * 256;
  for (; i < total; i += stride) {
    int idx512 = (int)(i & 511);
    long long chunk = i >> 9;
    int k32 = (int)(chunk % k32max);
    int nt  = (int)(chunk / k32max);
    int lane = idx512 >> 4, ii = idx512 & 15;
    int k = k32 * 32 + (lane >> 4) * 16 + ii;
    int n = nt * 16 + (lane & 15);
    dst[ebase + i] = f2bf(src[ebase + (size_t)k * N + n]);
  }
}

// ---------- LayerNorm: f32 [rows, DM] -> bf16 ----------
__global__ __launch_bounds__(256) void k_layernorm_bf16(const float* __restrict__ x,
    const float* __restrict__ g, const float* __restrict__ b, __bf16* __restrict__ out) {
  int row = blockIdx.x;
  const float* xr = x + (size_t)row * DM;
  __shared__ float rs[256], rq[256];
  float s = 0.f, s2 = 0.f;
  for (int c = threadIdx.x; c < DM; c += 256) { float v = xr[c]; s += v; s2 += v * v; }
  rs[threadIdx.x] = s; rq[threadIdx.x] = s2; __syncthreads();
  for (int o = 128; o > 0; o >>= 1) {
    if (threadIdx.x < o) { rs[threadIdx.x] += rs[threadIdx.x + o]; rq[threadIdx.x] += rq[threadIdx.x + o]; }
    __syncthreads();
  }
  float mu  = rs[0] * (1.0f / DM);
  float var = rq[0] * (1.0f / DM) - mu * mu;
  float rstd = rsqrtf(var + 1e-5f);
  __bf16* orow = out + (size_t)row * DM;
  for (int c = threadIdx.x; c < DM; c += 256)
    orow[c] = f2bf((xr[c] - mu) * rstd * g[c] + b[c]);
}

// ---------- WMMA block-GEMM mainloop (A row-major bf16, B pre-packed) ----------
// Block = 256 threads (8 waves). Block tile 128(M) x 128(N), K step 32.
// Wave w: mw=w>>1 (32-row subtile pair), nw=w&1 (64-col half); 2x4 accumulators.
// Double-buffered LDS; B staged by the Tensor Data Mover when available.
__device__ __forceinline__ void wmma_tile_gemm(
    const __bf16* __restrict__ A,     // [M x K] row-major, lda == K
    const __bf16* __restrict__ Bp,    // packed chunks
    int K, int m0, int mEnd, int n0, v8f acc[2][4]) {
  __shared__ __align__(32) __bf16 sA[2][128][32];
  __shared__ __align__(32) __bf16 sB[2][8 * 512];
  const int tid = threadIdx.x;
  const int lane = tid & 31, wave = tid >> 5;
  const int mw = wave >> 1, nw = wave & 1;
  const int half = lane >> 4, ln = lane & 15;
  const int k32max = K >> 5;
  const int nt_base = n0 >> 4;
  // staging maps: A thread -> (row, 16-col half); B thread -> (chunk, 16-elem slot)
  const int ar = tid >> 1, aseg = (tid & 1) * 16;
  const int bch = tid >> 5, bof = (tid & 31) * 16;
  const int agr = m0 + ar;
  const __bf16* aptr = A + (size_t)agr * K + aseg;
  const __bf16* bbase = Bp + ((size_t)(nt_base + bch) * k32max) * 512 + bof;

  bf16x8 pa0, pa1;
  auto load_tileA = [&](int k32) {
    pa0 = bf16x8{}; pa1 = bf16x8{};
    if (agr < mEnd) {
      const __bf16* ap = aptr + k32 * 32;
      pa0 = *(const bf16x8*)ap;
      pa1 = *(const bf16x8*)(ap + 8);
    }
  };
  auto store_tileA = [&](int buf) {
    *(bf16x8*)&sA[buf][ar][aseg]     = pa0;
    *(bf16x8*)&sA[buf][ar][aseg + 8] = pa1;
  };
#if USE_TDM
  const __bf16* btile0 = Bp + (size_t)nt_base * k32max * 512;  // chunk row 0
  const unsigned bstride = (unsigned)k32max * 1024u;           // bytes between chunk rows
  auto stage_b_issue = [&](int buf, int k32) {
    if (wave == 0)
      tdm_load_b((unsigned)(size_t)&sB[buf][0], btile0 + (size_t)k32 * 512, bstride);
  };
  auto stage_b_wait = [&](int) {
    if (wave == 0) __builtin_amdgcn_s_wait_tensorcnt(0);
  };
#else
  bf16x8 pb0, pb1;
  auto stage_b_issue = [&](int, int k32) {
    const __bf16* bp = bbase + (size_t)k32 * 512;
    pb0 = *(const bf16x8*)bp;
    pb1 = *(const bf16x8*)(bp + 8);
  };
  auto stage_b_wait = [&](int buf) {
    __bf16* d = sB[buf] + bch * 512 + bof;
    *(bf16x8*)d       = pb0;
    *(bf16x8*)(d + 8) = pb1;
  };
#endif

  load_tileA(0);
  stage_b_issue(0, 0);
  store_tileA(0);
  stage_b_wait(0);
  __syncthreads();
  int cur = 0;
  for (int k32 = 0; k32 < k32max; ++k32) {
    bool hasNext = (k32 + 1 < k32max);
    if (hasNext) {
      load_tileA(k32 + 1);              // overlap with WMMA below
      stage_b_issue(cur ^ 1, k32 + 1);  // DMA into the buffer nobody reads now
    }
    if (k32 + 4 < k32max)               // pull weight stream into L2
      __builtin_prefetch(bbase + (size_t)(k32 + 4) * 512, 0, 1);
    v16bf bfr[4];
#pragma unroll
    for (int t = 0; t < 4; ++t)
      bfr[t] = *(const v16bf*)(sB[cur] + (nw * 4 + t) * 512 + lane * 16);
#pragma unroll
    for (int mi = 0; mi < 2; ++mi) {
      int am = mw * 32 + mi * 16 + ln;
      v16bf a = frag2(&sA[cur][am][half * 16], &sA[cur][am][half * 16 + 8]);
#pragma unroll
      for (int t = 0; t < 4; ++t)
        acc[mi][t] = __builtin_amdgcn_wmma_f32_16x16x32_bf16(false, a, false, bfr[t], (short)0, acc[mi][t], false, false);
    }
    if (hasNext) {
      store_tileA(cur ^ 1);
      stage_b_wait(cur ^ 1);
      __syncthreads();
      cur ^= 1;
    }
  }
}

// epilogue index helpers: row = m0 + mw*32 + mi*16 + r + 8*half ; col = n0 + nw*64 + t*16 + ln

// ---------- GEMM, f32 output, optional bias + residual ----------
__global__ __launch_bounds__(256) void k_gemm_bf16_f32(
    const __bf16* __restrict__ A, const __bf16* __restrict__ Bp,
    const float* __restrict__ bias, const float* __restrict__ residual,
    float* __restrict__ out, int M, int N, int K) {
  int tilesN = N >> 7;
  int mt = blockIdx.x / tilesN, nt = blockIdx.x % tilesN;
  int m0 = mt * 128, n0 = nt * 128;
  v8f z = {};
  v8f acc[2][4];
#pragma unroll
  for (int mi = 0; mi < 2; ++mi)
#pragma unroll
    for (int t = 0; t < 4; ++t) acc[mi][t] = z;
  wmma_tile_gemm(A, Bp, K, m0, M, n0, acc);
  const int lane = threadIdx.x & 31, wave = threadIdx.x >> 5;
  const int mw = wave >> 1, nw = wave & 1, half = lane >> 4, ln = lane & 15;
#pragma unroll
  for (int mi = 0; mi < 2; ++mi)
#pragma unroll
    for (int t = 0; t < 4; ++t)
#pragma unroll
      for (int r = 0; r < 8; ++r) {
        int row = m0 + mw * 32 + mi * 16 + r + 8 * half;
        int col = n0 + nw * 64 + t * 16 + ln;
        if (row < M) {
          float v = acc[mi][t][r];
          if (bias) v += bias[col];
          if (residual) v += residual[(size_t)row * N + col];
          out[(size_t)row * N + col] = v;
        }
      }
}

// ---------- RoPE + 1/sqrt(hd) scale: qkv f32 -> Q/K/V bf16 [B*H, T, HD] ----------
__global__ __launch_bounds__(256) void k_rope(const float* __restrict__ qkv,
    __bf16* __restrict__ Q, __bf16* __restrict__ K, __bf16* __restrict__ V) {
  int idx = blockIdx.x * 256 + threadIdx.x;   // NTOK*NH*32 threads
  int i  = idx & 31;
  int h  = (idx >> 5) & (NH - 1);
  int tk = idx >> 9;
  int t  = tk & (TT - 1);
  int b  = tk >> 10;
  const float* row = qkv + (size_t)tk * (3 * DM);
  float inv = __powf(10000.0f, -(float)i * (1.0f / 32.0f));
  float ang = (float)t * inv;
  float cs = __cosf(ang), sn = __sinf(ang);
  float q1 = row[h * HD + i],            q2 = row[h * HD + i + 32];
  float k1 = row[DM + h * HD + i],       k2 = row[DM + h * HD + i + 32];
  float v1 = row[2 * DM + h * HD + i],   v2 = row[2 * DM + h * HD + i + 32];
  const float scale = 0.125f;            // HEAD_DIM^-0.5
  size_t base = (((size_t)(b * NH + h)) * TT + t) * HD;
  Q[base + i]      = f2bf((q1 * cs - q2 * sn) * scale);
  Q[base + i + 32] = f2bf((q2 * cs + q1 * sn) * scale);
  K[base + i]      = f2bf(k1 * cs - k2 * sn);
  K[base + i + 32] = f2bf(k2 * cs + k1 * sn);
  V[base + i]      = f2bf(v1);
  V[base + i + 32] = f2bf(v2);
}

// ---------- flash attention (causal), 64x64 tiles, WMMA QK^T and P*V ----------
__global__ __launch_bounds__(128) void k_flash_attn(const __bf16* __restrict__ Q,
    const __bf16* __restrict__ K, const __bf16* __restrict__ V, __bf16* __restrict__ out) {
  __shared__ __align__(32) __bf16 sQ[64][64];
  __shared__ __align__(32) __bf16 sK[64][64];
  __shared__ __align__(32) __bf16 sVt[64][64];   // transposed: [d][key]
  __shared__ __align__(32) __bf16 sP[64][64];
  __shared__ float  sS[64][65];
  __shared__ float  m_s[64], l_s[64], a_s[64];

  const int bh = blockIdx.y;           // b*NH + h
  const int qt = blockIdx.x;           // q tile (64 rows)
  const int tid = threadIdx.x, lane = tid & 31, wave = tid >> 5;
  const int half = lane >> 4, ln = lane & 15;
  const size_t qbase = ((size_t)bh * TT + (size_t)qt * 64) * HD;
  const v8f vzero = {};

  for (int i = tid; i < 64 * 64; i += 128) sQ[i >> 6][i & 63] = Q[qbase + i];
  if (tid < 64) { m_s[tid] = -1e30f; l_s[tid] = 0.f; }
  v8f acc[4];
#pragma unroll
  for (int t = 0; t < 4; ++t) acc[t] = vzero;
  __syncthreads();

  const int am = wave * 16 + ln;
  for (int j = 0; j <= qt; ++j) {
    size_t kbase = ((size_t)bh * TT + (size_t)j * 64) * HD;
    for (int i = tid; i < 64 * 64; i += 128) {
      sK[i >> 6][i & 63]  = K[kbase + i];
      sVt[i & 63][i >> 6] = V[kbase + i];   // transpose on store
    }
    __syncthreads();

    // S = Q K^T  (per wave: 16 q-rows x 64 k-cols)
    v8f sv[4];
#pragma unroll
    for (int t = 0; t < 4; ++t) sv[t] = vzero;
#pragma unroll
    for (int kk = 0; kk < 2; ++kk) {
      int kb = kk * 32;
      v16bf a = frag2(&sQ[am][kb + half * 16], &sQ[am][kb + half * 16 + 8]);
#pragma unroll
      for (int t = 0; t < 4; ++t) {
        v16bf b = *(const v16bf*)&sK[t * 16 + ln][kb + half * 16];
        sv[t] = __builtin_amdgcn_wmma_f32_16x16x32_bf16(false, a, false, b, (short)0, sv[t], false, false);
      }
    }
#pragma unroll
    for (int t = 0; t < 4; ++t)
#pragma unroll
      for (int r = 0; r < 8; ++r) {
        int rl = wave * 16 + r + 8 * half;
        int cl = t * 16 + ln;
        float v = sv[t][r];
        if (j * 64 + cl > qt * 64 + rl) v += -1e9f;   // causal additive mask
        sS[rl][cl] = v;
      }
    __syncthreads();

    // online softmax (one thread per q-row)
    if (tid < 64) {
      float mold = m_s[tid], mx = mold;
      for (int c = 0; c < 64; ++c) mx = fmaxf(mx, sS[tid][c]);
      float al = __expf(mold - mx);
      float sum = 0.f;
      for (int c = 0; c < 64; ++c) { float pv = __expf(sS[tid][c] - mx); sP[tid][c] = f2bf(pv); sum += pv; }
      l_s[tid] = l_s[tid] * al + sum;
      m_s[tid] = mx;
      a_s[tid] = al;
    }
    __syncthreads();

    // O = O*alpha + P V
#pragma unroll
    for (int t = 0; t < 4; ++t)
#pragma unroll
      for (int r = 0; r < 8; ++r) acc[t][r] *= a_s[wave * 16 + r + 8 * half];
#pragma unroll
    for (int kk = 0; kk < 2; ++kk) {
      int kb = kk * 32;
      v16bf a = frag2(&sP[am][kb + half * 16], &sP[am][kb + half * 16 + 8]);
#pragma unroll
      for (int t = 0; t < 4; ++t) {
        v16bf b = *(const v16bf*)&sVt[t * 16 + ln][kb + half * 16];
        acc[t] = __builtin_amdgcn_wmma_f32_16x16x32_bf16(false, a, false, b, (short)0, acc[t], false, false);
      }
    }
    __syncthreads();
  }

  // write to [token, d_model] bf16 layout
  int b = bh >> 4, h = bh & (NH - 1);
#pragma unroll
  for (int t = 0; t < 4; ++t)
#pragma unroll
    for (int r = 0; r < 8; ++r) {
      int rl = wave * 16 + r + 8 * half;
      int cl = t * 16 + ln;
      int tok = b * TT + qt * 64 + rl;
      out[(size_t)tok * DM + h * HD + cl] = f2bf(acc[t][r] / l_s[rl]);
    }
}

// ---------- router: logits, softmax, top-2, stats ----------
__global__ __launch_bounds__(256) void k_router(const __bf16* __restrict__ xln,
    const float* __restrict__ wr, int* __restrict__ top_i, float* __restrict__ top_w,
    float* __restrict__ prob_sums, int* __restrict__ counts) {
  int n = blockIdx.x;
  int lane = threadIdx.x & 31;
  int grp = threadIdx.x >> 5;              // expert 0..7, one wave each
  __shared__ float sl[8];
  const __bf16* xr = xln + (size_t)n * DM;
  float part = 0.f;
  for (int i = lane; i < DM; i += 32) part += bf2f(xr[i]) * wr[i * NE + grp];
#pragma unroll
  for (int o = 16; o > 0; o >>= 1) part += __shfl_xor(part, o, 32);
  if (lane == 0) sl[grp] = part;
  __syncthreads();
  if (threadIdx.x == 0) {
    float mx = sl[0];
    for (int e = 1; e < NE; ++e) mx = fmaxf(mx, sl[e]);
    float p[NE]; float sum = 0.f;
    for (int e = 0; e < NE; ++e) { p[e] = __expf(sl[e] - mx); sum += p[e]; }
    float inv = 1.0f / sum;
    for (int e = 0; e < NE; ++e) { p[e] *= inv; atomicAdd(&prob_sums[e], p[e]); }
    int i0 = 0; for (int e = 1; e < NE; ++e) if (p[e] > p[i0]) i0 = e;
    int i1 = (i0 == 0) ? 1 : 0;
    for (int e = 0; e < NE; ++e) if (e != i0 && p[e] > p[i1]) i1 = e;
    float ws = p[i0] + p[i1];
    top_i[n * 2 + 0] = i0; top_i[n * 2 + 1] = i1;
    top_w[n * 2 + 0] = p[i0] / ws; top_w[n * 2 + 1] = p[i1] / ws;
    atomicAdd(&counts[i0], 1); atomicAdd(&counts[i1], 1);
  }
}

__global__ void k_route_init(int* counts, int* cursors, float* psums) {
  int i = threadIdx.x;
  if (i < NE) { counts[i] = 0; cursors[i] = 0; psums[i] = 0.f; }
}

__global__ void k_route_finalize(const int* __restrict__ counts, const float* __restrict__ psums,
                                 int* __restrict__ offsets, float* __restrict__ aux_out) {
  if (threadIdx.x == 0) {
    int off = 0; float aux = 0.f;
    for (int e = 0; e < NE; ++e) {
      offsets[e] = off; off += counts[e];
      float frac = (float)counts[e] / (float)(NTOK * TOPK);
      float mp = psums[e] / (float)NTOK;
      aux += frac * mp;
    }
    *aux_out = (float)NE * aux;
  }
}

// ---------- gather tokens into expert-grouped rows ----------
__global__ __launch_bounds__(256) void k_gather(const __bf16* __restrict__ xln,
    const int* __restrict__ top_i, const float* __restrict__ top_w,
    const int* __restrict__ offsets, int* __restrict__ cursors,
    __bf16* __restrict__ gx, int* __restrict__ row_token, float* __restrict__ row_gate) {
  int idx = blockIdx.x;                 // 0..NASSIGN-1
  int n = idx >> 1;
  int e = top_i[idx];
  __shared__ int sslot;
  if (threadIdx.x == 0) sslot = atomicAdd(&cursors[e], 1);
  __syncthreads();
  int g = offsets[e] + sslot;
  const bf16x8* src = (const bf16x8*)(xln + (size_t)n * DM);
  bf16x8* dst = (bf16x8*)(gx + (size_t)g * DM);
  for (int c = threadIdx.x; c < DM / 8; c += 256) dst[c] = src[c];
  if (threadIdx.x == 0) { row_token[g] = n; row_gate[g] = top_w[idx]; }
}

// ---------- FFN1: h = gelu(gx @ w1[e] + b1[e]) -> bf16 ----------
__global__ __launch_bounds__(256) void k_gemm_ffn1(
    const __bf16* __restrict__ gx, const __bf16* __restrict__ w1p, const float* __restrict__ b1,
    const int* __restrict__ counts, const int* __restrict__ offsets, __bf16* __restrict__ h) {
  int e = blockIdx.y;
  int rows = counts[e];
  int tilesN = DFF >> 7;                // 32
  int mt = blockIdx.x / tilesN, nt = blockIdx.x % tilesN;
  int m0 = mt * 128;
  if (m0 >= rows) return;
  int n0 = nt * 128;
  const __bf16* A  = gx  + (size_t)offsets[e] * DM;
  const __bf16* Bp = w1p + (size_t)e * DM * DFF;
  v8f z = {}; v8f acc[2][4];
#pragma unroll
  for (int mi = 0; mi < 2; ++mi)
#pragma unroll
    for (int t = 0; t < 4; ++t) acc[mi][t] = z;
  wmma_tile_gemm(A, Bp, DM, m0, rows, n0, acc);
  const int lane = threadIdx.x & 31, wave = threadIdx.x >> 5;
  const int mw = wave >> 1, nw = wave & 1, half = lane >> 4, ln = lane & 15;
#pragma unroll
  for (int mi = 0; mi < 2; ++mi)
#pragma unroll
    for (int t = 0; t < 4; ++t)
#pragma unroll
      for (int r = 0; r < 8; ++r) {
        int row = m0 + mw * 32 + mi * 16 + r + 8 * half;
        int col = n0 + nw * 64 + t * 16 + ln;
        if (row < rows) {
          float v = acc[mi][t][r] + b1[e * DFF + col];
          float gel = 0.5f * v * (1.0f + erff(v * 0.70710678118654752f));
          h[((size_t)offsets[e] + row) * DFF + col] = f2bf(gel);
        }
      }
}

// ---------- FFN2: y = h @ w2[e] + b2[e]; scatter-add gate*y onto output ----------
__global__ __launch_bounds__(256) void k_gemm_ffn2(
    const __bf16* __restrict__ h, const __bf16* __restrict__ w2p, const float* __restrict__ b2,
    const int* __restrict__ counts, const int* __restrict__ offsets,
    const int* __restrict__ row_token, const float* __restrict__ row_gate,
    float* __restrict__ out) {
  int e = blockIdx.y;
  int rows = counts[e];
  int tilesN = DM >> 7;                 // 8
  int mt = blockIdx.x / tilesN, nt = blockIdx.x % tilesN;
  int m0 = mt * 128;
  if (m0 >= rows) return;
  int n0 = nt * 128;
  const __bf16* A  = h   + (size_t)offsets[e] * DFF;
  const __bf16* Bp = w2p + (size_t)e * DFF * DM;
  v8f z = {}; v8f acc[2][4];
#pragma unroll
  for (int mi = 0; mi < 2; ++mi)
#pragma unroll
    for (int t = 0; t < 4; ++t) acc[mi][t] = z;
  wmma_tile_gemm(A, Bp, DFF, m0, rows, n0, acc);
  const int lane = threadIdx.x & 31, wave = threadIdx.x >> 5;
  const int mw = wave >> 1, nw = wave & 1, half = lane >> 4, ln = lane & 15;
#pragma unroll
  for (int mi = 0; mi < 2; ++mi)
#pragma unroll
    for (int t = 0; t < 4; ++t)
#pragma unroll
      for (int r = 0; r < 8; ++r) {
        int row = m0 + mw * 32 + mi * 16 + r + 8 * half;
        int col = n0 + nw * 64 + t * 16 + ln;
        if (row < rows) {
          int g = offsets[e] + row;
          int tok = row_token[g];
          float gate = row_gate[g];
          float v = acc[mi][t][r] + b2[e * DM + col];
          atomicAdd(&out[(size_t)tok * DM + col], gate * v);
        }
      }
}

// ---------- host orchestration ----------
extern "C" void kernel_launch(void* const* d_in, const int* in_sizes, int n_in,
                              void* d_out, int out_size, void* d_ws, size_t ws_size,
                              hipStream_t stream) {
  const float* x      = (const float*)d_in[0];
  // d_in[1] = mask (causal mask applied analytically)
  const float* ln1_g  = (const float*)d_in[2];
  const float* ln1_b  = (const float*)d_in[3];
  const float* w_qkv  = (const float*)d_in[4];
  const float* w_proj = (const float*)d_in[5];
  const float* b_proj = (const float*)d_in[6];
  const float* ln2_g  = (const float*)d_in[7];
  const float* ln2_b  = (const float*)d_in[8];
  const float* w_rout = (const float*)d_in[9];
  const float* w1     = (const float*)d_in[10];
  const float* b1     = (const float*)d_in[11];
  const float* w2     = (const float*)d_in[12];
  const float* b2     = (const float*)d_in[13];
  float* out = (float*)d_out;

  char* base = (char*)d_ws;
  size_t off = 0;
  auto take = [&](size_t bytes) -> char* {
    char* r = base + off;
    off += (bytes + 255) & ~(size_t)255;
    return r;
  };

  __bf16* xln1   = (__bf16*)take((size_t)NTOK * DM * 2);
  __bf16* wqkvp  = (__bf16*)take((size_t)DM * 3 * DM * 2);
  float*  qkv    = (float*) take((size_t)NTOK * 3 * DM * 4);
  __bf16* Qb     = (__bf16*)take((size_t)NTOK * DM * 2);
  __bf16* Kb     = (__bf16*)take((size_t)NTOK * DM * 2);
  __bf16* Vb     = (__bf16*)take((size_t)NTOK * DM * 2);
  __bf16* attnb  = (__bf16*)take((size_t)NTOK * DM * 2);
  __bf16* wprojp = (__bf16*)take((size_t)DM * DM * 2);
  __bf16* xln2   = (__bf16*)take((size_t)NTOK * DM * 2);
  __bf16* w1p    = (__bf16*)take((size_t)NE * DM * DFF * 2);
  __bf16* w2p    = (__bf16*)take((size_t)NE * DFF * DM * 2);
  __bf16* gx     = (__bf16*)take((size_t)NASSIGN * DM * 2);
  __bf16* hbuf   = (__bf16*)take((size_t)NASSIGN * DFF * 2);
  int*    top_i  = (int*)  take((size_t)NASSIGN * 4);
  float*  top_w  = (float*)take((size_t)NASSIGN * 4);
  int*    row_tk = (int*)  take((size_t)NASSIGN * 4);
  float*  row_g  = (float*)take((size_t)NASSIGN * 4);
  int*    counts = (int*)  take(64);
  int*    offs   = (int*)  take(64);
  int*    cursor = (int*)  take(64);
  float*  psums  = (float*)take(64);

  // pack weights into bf16 WMMA-B fragment layout (one-time per call)
  k_pack_b<<<dim3(1024, 1),  256, 0, stream>>>(w_qkv,  wqkvp,  DM,  3 * DM);
  k_pack_b<<<dim3(512, 1),   256, 0, stream>>>(w_proj, wprojp, DM,  DM);
  k_pack_b<<<dim3(2048, NE), 256, 0, stream>>>(w1,     w1p,    DM,  DFF);
  k_pack_b<<<dim3(2048, NE), 256, 0, stream>>>(w2,     w2p,    DFF, DM);

  // attention path
  k_layernorm_bf16<<<NTOK, 256, 0, stream>>>(x, ln1_g, ln1_b, xln1);
  k_gemm_bf16_f32<<<(NTOK / 128) * (3 * DM / 128), 256, 0, stream>>>(
      xln1, wqkvp, nullptr, nullptr, qkv, NTOK, 3 * DM, DM);
  k_rope<<<(NTOK * NH * 32) / 256, 256, 0, stream>>>(qkv, Qb, Kb, Vb);
  k_flash_attn<<<dim3(TT / 64, BB * NH), 128, 0, stream>>>(Qb, Kb, Vb, attnb);
  k_gemm_bf16_f32<<<(NTOK / 128) * (DM / 128), 256, 0, stream>>>(
      attnb, wprojp, b_proj, x, out, NTOK, DM, DM);   // out = x + attn_proj

  // MoE path (sparse top-2 routing)
  k_layernorm_bf16<<<NTOK, 256, 0, stream>>>(out, ln2_g, ln2_b, xln2);
  k_route_init<<<1, 32, 0, stream>>>(counts, cursor, psums);
  k_router<<<NTOK, 256, 0, stream>>>(xln2, w_rout, top_i, top_w, psums, counts);
  k_route_finalize<<<1, 32, 0, stream>>>(counts, psums, offs, out + (size_t)NTOK * DM);
  k_gather<<<NASSIGN, 256, 0, stream>>>(xln2, top_i, top_w, offs, cursor, gx, row_tk, row_g);
  k_gemm_ffn1<<<dim3((NTOK / 128) * (DFF / 128), NE), 256, 0, stream>>>(
      gx, w1p, b1, counts, offs, hbuf);
  k_gemm_ffn2<<<dim3((NTOK / 128) * (DM / 128), NE), 256, 0, stream>>>(
      hbuf, w2p, b2, counts, offs, row_tk, row_g, out);

  (void)in_sizes; (void)n_in; (void)out_size; (void)ws_size;
}
